// MultiHeadSelfAttention_30709016166727
// MI455X (gfx1250) — compile-verified, matrix operand order checked by reading, not run on hardware
//
#include <hip/hip_runtime.h>

typedef __attribute__((ext_vector_type(16))) __bf16 v16bf;
typedef __attribute__((ext_vector_type(8)))  __bf16 v8bf;
typedef __attribute__((ext_vector_type(4)))  __bf16 v4bf;
typedef __attribute__((ext_vector_type(8)))  float  v8f;
typedef __attribute__((ext_vector_type(4)))  float  v4f;

#define D_MODEL 1024
#define NUM_HEADS 16
#define DK 64
#define SEQ 2048
#define BATCH 4
#define MROWS (BATCH * SEQ)  // 8192

// ---------------------------------------------------------------------------
// Fragment loader for 16x16x32 bf16 WMMA (A and B share the per-lane pattern
// on CDNA5): lane L reads row (L&15); k-offsets (L>>4)*8..+7 and +16..+23.
// Works for both global and LDS tiles (compiler infers address space).
// ---------------------------------------------------------------------------
__device__ __forceinline__ v16bf load_frag(const __bf16* __restrict__ tile,
                                           int stride, int lane) {
  const __bf16* row = tile + (size_t)(lane & 15) * stride + ((lane >> 4) << 3);
  v8bf lo = *reinterpret_cast<const v8bf*>(row);
  v8bf hi = *reinterpret_cast<const v8bf*>(row + 16);
  v16bf a;
#pragma unroll
  for (int i = 0; i < 8; ++i) { a[i] = lo[i]; a[i + 8] = hi[i]; }
  return a;
}

__device__ __forceinline__ v8f wmma_bf16(v16bf a, v16bf b, v8f c) {
  return __builtin_amdgcn_wmma_f32_16x16x32_bf16(
      false, a, false, b, (short)0, c, false, false);
}

// Async 16B copy global -> LDS (per-lane). VDST = LDS byte offset (low 32 bits
// of the flat shared-aperture address), VADDR = 64-bit global address.
__device__ __forceinline__ void async_b128(const __bf16* gsrc,
                                           __bf16* lds_dst) {
  unsigned lds_off = (unsigned)(uintptr_t)lds_dst;
  asm volatile("global_load_async_to_lds_b128 %0, %1, off"
               :: "v"(lds_off), "v"(gsrc)
               : "memory");
}

// Stage hand-off barrier: drain only ASYNCcnt (our LDS writes) and DScnt (our
// LDS reads) -- NOT LOADcnt, so register-prefetched B-frag global loads keep
// flying across the barrier.
__device__ __forceinline__ void sync_stage() {
  asm volatile(
      "s_wait_asynccnt 0x0\n\t"
      "s_wait_dscnt 0x0\n\t"
      "s_barrier_signal -1\n\t"
      "s_barrier_wait -1"
      ::: "memory");
}

// ---------------------------------------------------------------------------
// fp32 -> bf16 cast, 4 elements/lane (b128 load, b64 store)
// ---------------------------------------------------------------------------
__global__ void cvt_f32_bf16(const float* __restrict__ src,
                             __bf16* __restrict__ dst, int n4) {
  int i = blockIdx.x * blockDim.x + threadIdx.x;
  const int stride = gridDim.x * blockDim.x;
  const v4f* s4 = reinterpret_cast<const v4f*>(src);
  v4bf* d4 = reinterpret_cast<v4bf*>(dst);
  for (; i < n4; i += stride) {
    v4f f = s4[i];
    v4bf o;
#pragma unroll
    for (int j = 0; j < 4; ++j) o[j] = (__bf16)f[j];
    d4[i] = o;
  }
}

// ---------------------------------------------------------------------------
// QKV projection: y[m,o] = sum_i xb[m,i] * W[o,i]   (y = x @ W^T)
// Block: 4 waves share a 32-row A-tile staged in LDS via async copies
// (double buffered); each wave owns a 64-col (one-head) B strip in registers.
// Q,K stored [b,h,s,dk]; V stored transposed [b,h,dk,s].
// ---------------------------------------------------------------------------
__global__ __launch_bounds__(128) void qkv_proj(
    const __bf16* __restrict__ xb, const __bf16* __restrict__ wq,
    const __bf16* __restrict__ wk, const __bf16* __restrict__ wv,
    __bf16* __restrict__ Qb, __bf16* __restrict__ Kb,
    __bf16* __restrict__ Vt) {
  __shared__ __align__(16) __bf16 At[2][32 * 32];  // 2 x 2KB double buffer
  const int tid   = threadIdx.x;
  const int lane  = tid & 31;
  const int wave  = tid >> 5;
  const int mbase = blockIdx.x * 32;
  const int obase = blockIdx.y * 256 + wave * 64;  // head-aligned 64 columns
  const int which = blockIdx.z;                    // 0=Q 1=K 2=V
  const __bf16* W = (which == 0) ? wq : (which == 1) ? wk : wv;

  // per-thread async element: row 0..31, 16B chunk 0..3 within the 32-col tile
  const int arow = tid >> 2;
  const int acol = (tid & 3) * 8;
  const __bf16* Asrc = xb + (size_t)(mbase + arow) * D_MODEL + acol;

  const __bf16* Br0 = W + (size_t)(obase +  0) * D_MODEL;
  const __bf16* Br1 = W + (size_t)(obase + 16) * D_MODEL;
  const __bf16* Br2 = W + (size_t)(obase + 32) * D_MODEL;
  const __bf16* Br3 = W + (size_t)(obase + 48) * D_MODEL;

  v8f acc[2][4] = {};

  // prologue: stage A(k=0), load B(k=0) into registers
  async_b128(Asrc, &At[0][arow * 32 + acol]);
  v16bf b0 = load_frag(Br0, D_MODEL, lane);
  v16bf b1 = load_frag(Br1, D_MODEL, lane);
  v16bf b2 = load_frag(Br2, D_MODEL, lane);
  v16bf b3 = load_frag(Br3, D_MODEL, lane);
  sync_stage();

  int buf = 0;
  for (int k = 0; k < D_MODEL; k += 32) {
    const int kn = (k + 32) % D_MODEL;  // branchless prefetch (wraps on last)
    async_b128(Asrc + kn, &At[buf ^ 1][arow * 32 + acol]);
    v16bf nb0 = load_frag(Br0 + kn, D_MODEL, lane);
    v16bf nb1 = load_frag(Br1 + kn, D_MODEL, lane);
    v16bf nb2 = load_frag(Br2 + kn, D_MODEL, lane);
    v16bf nb3 = load_frag(Br3 + kn, D_MODEL, lane);

    v16bf a0 = load_frag(&At[buf][0],       32, lane);  // rows 0..15
    v16bf a1 = load_frag(&At[buf][16 * 32], 32, lane);  // rows 16..31

    acc[0][0] = wmma_bf16(a0, b0, acc[0][0]);
    acc[1][0] = wmma_bf16(a1, b0, acc[1][0]);
    acc[0][1] = wmma_bf16(a0, b1, acc[0][1]);
    acc[1][1] = wmma_bf16(a1, b1, acc[1][1]);
    acc[0][2] = wmma_bf16(a0, b2, acc[0][2]);
    acc[1][2] = wmma_bf16(a1, b2, acc[1][2]);
    acc[0][3] = wmma_bf16(a0, b3, acc[0][3]);
    acc[1][3] = wmma_bf16(a1, b3, acc[1][3]);

    sync_stage();  // async writes landed + LDS reads done; LOADcnt untouched
    buf ^= 1;
    b0 = nb0; b1 = nb1; b2 = nb2; b3 = nb3;
  }

  const int b  = mbase / SEQ;
  const int h  = obase / DK;
  const int n  = lane & 15;
  const size_t bh = (size_t)b * NUM_HEADS + h;
#pragma unroll
  for (int half = 0; half < 2; ++half) {
    const int s0 = (mbase % SEQ) + half * 16;
#pragma unroll
    for (int dc = 0; dc < 4; ++dc) {
      const int d = dc * 16 + n;
#pragma unroll
      for (int r = 0; r < 8; ++r) {
        const int s = s0 + r + ((lane >> 4) << 3);
        __bf16 v = (__bf16)acc[half][dc][r];
        if (which == 0)      Qb[(bh * SEQ + s) * DK + d] = v;
        else if (which == 1) Kb[(bh * SEQ + s) * DK + d] = v;
        else                 Vt[(bh * DK + d) * SEQ + s] = v;
      }
    }
  }
}

// ---------------------------------------------------------------------------
// Causal flash attention. One wave per 16-row q-tile; 32-key tiles.
// K-frags for next tile + V-frags issued before softmax so VALU hides vmem.
// ---------------------------------------------------------------------------
__global__ __launch_bounds__(128) void attn_fwd(
    const __bf16* __restrict__ Qb, const __bf16* __restrict__ Kb,
    const __bf16* __restrict__ Vt, __bf16* __restrict__ Ob) {
  __shared__ __align__(16) __bf16 lds[4][16 * 32];
  const int lane = threadIdx.x & 31;
  const int wave = threadIdx.x >> 5;
  const int wid  = blockIdx.x * 4 + wave;  // 0..8191
  const int qt = wid & 127;
  const int h  = (wid >> 7) & 15;
  const int b  = wid >> 11;
  const size_t bh = (size_t)b * NUM_HEADS + h;
  const __bf16* Qh = Qb + bh * SEQ * DK;
  const __bf16* Kh = Kb + bh * SEQ * DK;
  const __bf16* Vh = Vt + bh * DK * SEQ;
  const int q0 = qt * 16;
  const float NEG_INF = -__builtin_inff();

  v16bf qa0 = load_frag(Qh + (size_t)q0 * DK + 0,  DK, lane);
  v16bf qa1 = load_frag(Qh + (size_t)q0 * DK + 32, DK, lane);

  float m_i[8], l_i[8];
#pragma unroll
  for (int r = 0; r < 8; ++r) { m_i[r] = NEG_INF; l_i[r] = 0.0f; }
  v8f acc[4] = {};

  const int nk = q0 + 16;  // keys visible to this q-tile

  v16bf kf0 = load_frag(Kh + 0,             DK, lane);
  v16bf kf1 = load_frag(Kh + 32,            DK, lane);
  v16bf kf2 = load_frag(Kh + (size_t)16 * DK + 0,  DK, lane);
  v16bf kf3 = load_frag(Kh + (size_t)16 * DK + 32, DK, lane);

  for (int kb = 0; kb < nk; kb += 32) {
    // ---- scores: 16 q-rows x 32 keys --------------------------------------
    v8f s0 = {}, s1 = {};
    s0 = wmma_bf16(qa0, kf0, s0);
    s0 = wmma_bf16(qa1, kf1, s0);
    s1 = wmma_bf16(qa0, kf2, s1);
    s1 = wmma_bf16(qa1, kf3, s1);

    // ---- issue V-frags (this tile) + K-frags (next tile) early ------------
    v16bf vf0 = load_frag(Vh + (size_t) 0 * SEQ + kb, SEQ, lane);
    v16bf vf1 = load_frag(Vh + (size_t)16 * SEQ + kb, SEQ, lane);
    v16bf vf2 = load_frag(Vh + (size_t)32 * SEQ + kb, SEQ, lane);
    v16bf vf3 = load_frag(Vh + (size_t)48 * SEQ + kb, SEQ, lane);
    const int kload = (kb + 32 < nk) ? (kb + 32) : kb;  // dummy reload on last
    v16bf nk0 = load_frag(Kh + (size_t)(kload)      * DK + 0,  DK, lane);
    v16bf nk1 = load_frag(Kh + (size_t)(kload)      * DK + 32, DK, lane);
    v16bf nk2 = load_frag(Kh + (size_t)(kload + 16) * DK + 0,  DK, lane);
    v16bf nk3 = load_frag(Kh + (size_t)(kload + 16) * DK + 32, DK, lane);

    // ---- online softmax (scale, causal mask, row max/sum) -----------------
    const int k0i = kb + (lane & 15);
    const int k1i = kb + 16 + (lane & 15);
    float p0[8], p1[8];
#pragma unroll
    for (int r = 0; r < 8; ++r) {
      const int q = q0 + r + ((lane >> 4) << 3);
      float x0 = s0[r] * 0.125f;  if (k0i > q) x0 = NEG_INF;
      float x1 = s1[r] * 0.125f;  if (k1i > q) x1 = NEG_INF;
      float t = fmaxf(x0, x1);
#pragma unroll
      for (int msk = 1; msk < 16; msk <<= 1)
        t = fmaxf(t, __shfl_xor(t, msk, 32));
      const float mnew  = fmaxf(m_i[r], t);
      const float alpha = __expf(m_i[r] - mnew);
      m_i[r] = mnew;
      const float e0 = __expf(x0 - mnew);
      const float e1 = __expf(x1 - mnew);
      float rs = e0 + e1;
#pragma unroll
      for (int msk = 1; msk < 16; msk <<= 1)
        rs += __shfl_xor(rs, msk, 32);
      l_i[r] = l_i[r] * alpha + rs;
#pragma unroll
      for (int dc = 0; dc < 4; ++dc) acc[dc][r] *= alpha;
      p0[r] = e0; p1[r] = e1;
    }

    // ---- transpose P (C/D layout -> A layout) through LDS -----------------
    __bf16* pl = lds[wave];
#pragma unroll
    for (int r = 0; r < 8; ++r) {
      const int row = r + ((lane >> 4) << 3);
      pl[row * 32 + (lane & 15)]      = (__bf16)p0[r];
      pl[row * 32 + 16 + (lane & 15)] = (__bf16)p1[r];
    }
    asm volatile("s_wait_dscnt 0" ::: "memory");  // LDS store->load hazard
    v16bf pa = load_frag(pl, 32, lane);

    // ---- P @ V ------------------------------------------------------------
    acc[0] = wmma_bf16(pa, vf0, acc[0]);
    acc[1] = wmma_bf16(pa, vf1, acc[1]);
    acc[2] = wmma_bf16(pa, vf2, acc[2]);
    acc[3] = wmma_bf16(pa, vf3, acc[3]);

    kf0 = nk0; kf1 = nk1; kf2 = nk2; kf3 = nk3;
    asm volatile("" ::: "memory");  // keep next-iter LDS stores behind loads
  }

  // ---- finalize: divide by row sum, store [b,s,h*64+d] as bf16 ------------
  const int n = lane & 15;
#pragma unroll
  for (int r = 0; r < 8; ++r) {
    const float inv = 1.0f / l_i[r];
    const int s = q0 + r + ((lane >> 4) << 3);
    const size_t rowoff = ((size_t)b * SEQ + s) * D_MODEL + h * DK;
#pragma unroll
    for (int dc = 0; dc < 4; ++dc)
      Ob[rowoff + dc * 16 + n] = (__bf16)(acc[dc][r] * inv);
  }
}

// ---------------------------------------------------------------------------
// Output projection: out[m,o] = sum_i Ob[m,i] * wo[o,i], fp32 result.
// Same async-LDS A-tile + register-buffered B scheme as qkv_proj.
// ---------------------------------------------------------------------------
__global__ __launch_bounds__(128) void out_proj(
    const __bf16* __restrict__ Ob, const __bf16* __restrict__ wo,
    float* __restrict__ out) {
  __shared__ __align__(16) __bf16 At[2][32 * 32];
  const int tid   = threadIdx.x;
  const int lane  = tid & 31;
  const int wave  = tid >> 5;
  const int mbase = blockIdx.x * 32;
  const int obase = blockIdx.y * 256 + wave * 64;

  const int arow = tid >> 2;
  const int acol = (tid & 3) * 8;
  const __bf16* Asrc = Ob + (size_t)(mbase + arow) * D_MODEL + acol;

  const __bf16* Br0 = wo + (size_t)(obase +  0) * D_MODEL;
  const __bf16* Br1 = wo + (size_t)(obase + 16) * D_MODEL;
  const __bf16* Br2 = wo + (size_t)(obase + 32) * D_MODEL;
  const __bf16* Br3 = wo + (size_t)(obase + 48) * D_MODEL;

  v8f acc[2][4] = {};

  async_b128(Asrc, &At[0][arow * 32 + acol]);
  v16bf b0 = load_frag(Br0, D_MODEL, lane);
  v16bf b1 = load_frag(Br1, D_MODEL, lane);
  v16bf b2 = load_frag(Br2, D_MODEL, lane);
  v16bf b3 = load_frag(Br3, D_MODEL, lane);
  sync_stage();

  int buf = 0;
  for (int k = 0; k < D_MODEL; k += 32) {
    const int kn = (k + 32) % D_MODEL;
    async_b128(Asrc + kn, &At[buf ^ 1][arow * 32 + acol]);
    v16bf nb0 = load_frag(Br0 + kn, D_MODEL, lane);
    v16bf nb1 = load_frag(Br1 + kn, D_MODEL, lane);
    v16bf nb2 = load_frag(Br2 + kn, D_MODEL, lane);
    v16bf nb3 = load_frag(Br3 + kn, D_MODEL, lane);

    v16bf a0 = load_frag(&At[buf][0],       32, lane);
    v16bf a1 = load_frag(&At[buf][16 * 32], 32, lane);

    acc[0][0] = wmma_bf16(a0, b0, acc[0][0]);
    acc[1][0] = wmma_bf16(a1, b0, acc[1][0]);
    acc[0][1] = wmma_bf16(a0, b1, acc[0][1]);
    acc[1][1] = wmma_bf16(a1, b1, acc[1][1]);
    acc[0][2] = wmma_bf16(a0, b2, acc[0][2]);
    acc[1][2] = wmma_bf16(a1, b2, acc[1][2]);
    acc[0][3] = wmma_bf16(a0, b3, acc[0][3]);
    acc[1][3] = wmma_bf16(a1, b3, acc[1][3]);

    sync_stage();
    buf ^= 1;
    b0 = nb0; b1 = nb1; b2 = nb2; b3 = nb3;
  }

  const int n = lane & 15;
#pragma unroll
  for (int half = 0; half < 2; ++half)
#pragma unroll
    for (int dc = 0; dc < 4; ++dc)
#pragma unroll
      for (int r = 0; r < 8; ++r) {
        const int m = mbase + half * 16 + r + ((lane >> 4) << 3);
        out[(size_t)m * D_MODEL + obase + dc * 16 + n] = acc[half][dc][r];
      }
}

// ---------------------------------------------------------------------------
extern "C" void kernel_launch(void* const* d_in, const int* in_sizes, int n_in,
                              void* d_out, int out_size, void* d_ws,
                              size_t ws_size, hipStream_t stream) {
  const float* x  = (const float*)d_in[0];
  const float* wq = (const float*)d_in[1];
  const float* wk = (const float*)d_in[2];
  const float* wv = (const float*)d_in[3];
  const float* wo = (const float*)d_in[4];
  float* out = (float*)d_out;

  __bf16* xb  = (__bf16*)d_ws;
  __bf16* wqb = xb  + (size_t)MROWS * D_MODEL;
  __bf16* wkb = wqb + (size_t)D_MODEL * D_MODEL;
  __bf16* wvb = wkb + (size_t)D_MODEL * D_MODEL;
  __bf16* wob = wvb + (size_t)D_MODEL * D_MODEL;
  __bf16* Qb  = wob + (size_t)D_MODEL * D_MODEL;   // [b,h,s,dk]
  __bf16* Kb  = Qb  + (size_t)MROWS * D_MODEL;
  __bf16* Vt  = Kb  + (size_t)MROWS * D_MODEL;     // [b,h,dk,s]
  __bf16* Ob  = Vt  + (size_t)MROWS * D_MODEL;     // [b,s,h*dk]

  cvt_f32_bf16<<<256, 256, 0, stream>>>(x, xb, MROWS * D_MODEL / 4);
  cvt_f32_bf16<<<32, 256, 0, stream>>>(wq, wqb, D_MODEL * D_MODEL / 4);
  cvt_f32_bf16<<<32, 256, 0, stream>>>(wk, wkb, D_MODEL * D_MODEL / 4);
  cvt_f32_bf16<<<32, 256, 0, stream>>>(wv, wvb, D_MODEL * D_MODEL / 4);
  cvt_f32_bf16<<<32, 256, 0, stream>>>(wo, wob, D_MODEL * D_MODEL / 4);

  dim3 gproj(MROWS / 32, D_MODEL / 256, 3);
  qkv_proj<<<gproj, 128, 0, stream>>>(xb, wqb, wkb, wvb, Qb, Kb, Vt);

  attn_fwd<<<(BATCH * NUM_HEADS * (SEQ / 16)) / 4, 128, 0, stream>>>(Qb, Kb,
                                                                     Vt, Ob);

  dim3 gout(MROWS / 32, D_MODEL / 256);
  out_proj<<<gout, 128, 0, stream>>>(Ob, wob, out);
}
